// BahdanauAttention_71416716198217
// MI455X (gfx1250) — matrix operand order repaired and checked by reading, not compiled, not run on hardware
//
#include <hip/hip_runtime.h>
#include <math.h>

// Problem constants (match the reference)
#define BB 32
#define TT 2048
#define DD 512
#define UU 512

typedef __attribute__((ext_vector_type(16))) __bf16 v16bf;
typedef __attribute__((ext_vector_type(8)))  float  v8f;
typedef __attribute__((__vector_size__(16))) int    v4i_vs;   // matches builtin param

#if defined(__has_builtin)
#if __has_builtin(__builtin_amdgcn_global_load_async_to_lds_b128)
#define USE_ASYNC 1
#endif
#endif
#ifndef USE_ASYNC
#define USE_ASYNC 0
#endif

__device__ __forceinline__ unsigned short f2bf(float f) {
    unsigned int u = __float_as_uint(f);
    unsigned int r = u + 0x7FFFu + ((u >> 16) & 1u);   // round-to-nearest-even
    return (unsigned short)(r >> 16);
}

// 16-byte global -> LDS copy (async path if available, else via VGPRs)
__device__ __forceinline__ void cp16_g2l(unsigned short* dst, const unsigned short* src) {
#if USE_ASYNC
    __builtin_amdgcn_global_load_async_to_lds_b128(
        (__attribute__((address_space(1))) v4i_vs*)src,
        (__attribute__((address_space(3))) v4i_vs*)dst,
        0, 0);
#else
    *(uint4*)dst = *(const uint4*)src;
#endif
}

__device__ __forceinline__ void cp_fence() {
#if USE_ASYNC
#if __has_builtin(__builtin_amdgcn_s_wait_asynccnt)
    __builtin_amdgcn_s_wait_asynccnt(0);
#else
    asm volatile("s_wait_asynccnt 0x0" ::: "memory");
#endif
#endif
    __syncthreads();
}

// ---------------------------------------------------------------------------
// Prep 1: W1t[u][d] = bf16(W1[d][u])   (so B fragments are contiguous loads)
// ---------------------------------------------------------------------------
__global__ void bahdanau_w1t_kernel(const float* __restrict__ W1,
                                    unsigned short* __restrict__ W1t) {
    int idx = blockIdx.x * 256 + threadIdx.x;      // U*D threads
    int u = idx / DD, d = idx % DD;
    W1t[idx] = f2bf(W1[d * UU + u]);
}

// ---------------------------------------------------------------------------
// Prep 2: projq[b][u] = sum_d query[b][d]*W2[d][u] + b2[u] + b1[u]  (fp32)
// ---------------------------------------------------------------------------
__global__ void bahdanau_projq_kernel(const float* __restrict__ query,
                                      const float* __restrict__ W2,
                                      const float* __restrict__ b1,
                                      const float* __restrict__ b2,
                                      float* __restrict__ projq) {
    int idx = blockIdx.x * 256 + threadIdx.x;      // B*U threads
    int b = idx / UU, u = idx % UU;
    float acc = b1[u] + b2[u];
    const float* q = query + b * DD;
    for (int d = 0; d < DD; ++d) acc += q[d] * W2[d * UU + u];
    projq[idx] = acc;
}

// ---------------------------------------------------------------------------
// Main fused kernel:
//   score[b][t] = sum_u tanh( (values@W1)[b,t,u] + projq[b,u] ) * V[u] + bV
// 8 waves / WG; each wave: 16 T-rows x full U=512, A resident in 128 VGPRs.
// W1t u-tiles (16KB) are staged into LDS once per WG, double-buffered with
// async global->LDS copies overlapped with the 16 WMMAs of the current tile.
// ---------------------------------------------------------------------------
__global__ __launch_bounds__(256, 1)
void bahdanau_score_kernel(const float* __restrict__ values,          // [B,T,D]
                           const unsigned short* __restrict__ W1t,    // [U,D] bf16
                           const float* __restrict__ projq,           // [B,U]
                           const float* __restrict__ V,               // [U]
                           const float* __restrict__ bV,              // [1]
                           float* __restrict__ score_out)             // [B,T]
{
    __shared__ unsigned short btile[2][16 * DD];   // 2 x 16KB double buffer

    const int tid  = threadIdx.x;
    const int b    = blockIdx.y;
    const int wave = tid >> 5;                   // 0..7
    const int lane = tid & 31;
    const int row  = lane & 15;                  // M (and N for B/C operands)
    const int hi   = lane >> 4;                  // lane half selects K group
    const int t0   = blockIdx.x * 128 + wave * 16;

    const float* vrow = values + ((size_t)b * TT + (t0 + row)) * DD;

    // --- Kick off staging of u-tile 0 into LDS buffer 0 ---------------------
    {
        const unsigned short* src = W1t;
        unsigned short* dst = &btile[0][0];
        #pragma unroll
        for (int p = 0; p < 4; ++p) {
            const int off = (p * 256 + tid) * 8;   // 16B chunks, coalesced
            cp16_g2l(dst + off, src + off);
        }
    }

    // --- Load entire A operand (16 rows x K=512) as 16 bf16 fragments -------
    // A k order chosen here (contiguous): lane half hi -> K base offset hi*16; elements 0..15
    // hold K = k0..k0+15 consecutive (matches B).
    v16bf afrag[16];
    #pragma unroll
    for (int ks = 0; ks < 16; ++ks) {
        const int k0 = ks * 32 + hi * 16;
        union { v16bf v; unsigned short s[16]; } au;
        #pragma unroll
        for (int e = 0; e < 8; ++e) {
            au.s[e]     = f2bf(vrow[k0 + e]);
            au.s[e + 8] = f2bf(vrow[k0 + 8 + e]);
        }
        afrag[ks] = au.v;
    }

    float partial[8];
    #pragma unroll
    for (int j = 0; j < 8; ++j) partial[j] = 0.0f;

    cp_fence();                                  // u-tile 0 resident in LDS

    // --- Sweep the 32 u-tiles; full-K accumulate then tanh/dot epilogue -----
    for (int ut = 0; ut < 32; ++ut) {
        const int cur = ut & 1;

        // Prefetch next u-tile into the other buffer (async, overlapped)
        if (ut + 1 < 32) {
            const unsigned short* src = W1t + (size_t)(ut + 1) * (16 * DD);
            unsigned short* dst = &btile[cur ^ 1][0];
            #pragma unroll
            for (int p = 0; p < 4; ++p) {
                const int off = (p * 256 + tid) * 8;
                cp16_g2l(dst + off, src + off);
            }
        }

        const int n = ut * 16 + row;             // output column / u index
        // B k order chosen here (same as A): lane holds column n, K = hi*16 + 0..15 consecutive.
        const unsigned short* wrow = &btile[cur][(size_t)row * DD + hi * 16];

        v8f c = {0.f, 0.f, 0.f, 0.f, 0.f, 0.f, 0.f, 0.f};
        #pragma unroll
        for (int ks = 0; ks < 16; ++ks) {
            v16bf bfrag = *(const v16bf*)(wrow + ks * 32);   // 2x ds_load_b128
            c = __builtin_amdgcn_wmma_f32_16x16x32_bf16(
                    false, afrag[ks], false, bfrag, (short)0, c, false, false);
        }

        const float pq = projq[(size_t)b * UU + n];
        const float vn = V[n];
        // C layout: VGPR j -> row (hi*8 + j), column n (= lane%16)
        #pragma unroll
        for (int j = 0; j < 8; ++j)
            partial[j] += tanhf(c[j] + pq) * vn;

        if (ut + 1 < 32) cp_fence();             // next tile ready, buf safe
    }

    // --- Reduce columns across each 16-lane half (rows j and 8+j) -----------
    #pragma unroll
    for (int j = 0; j < 8; ++j) {
        float v = partial[j];
        #pragma unroll
        for (int off = 1; off < 16; off <<= 1)
            v += __shfl_xor(v, off, 32);
        partial[j] = v;
    }

    const float bv = bV[0];
    if (row == 0) {                              // lane 0 -> rows 0..7, lane 16 -> rows 8..15
        #pragma unroll
        for (int j = 0; j < 8; ++j)
            score_out[(size_t)b * TT + t0 + hi * 8 + j] = partial[j] + bv;
    }
}

// ---------------------------------------------------------------------------
// Softmax over T per batch (in place on the attention output region)
// ---------------------------------------------------------------------------
__global__ __launch_bounds__(256)
void bahdanau_softmax_kernel(float* __restrict__ attn) {
    __shared__ float red[256];
    const int b = blockIdx.x, tid = threadIdx.x;
    float* s = attn + (size_t)b * TT;

    float m = -INFINITY;
    for (int t = tid; t < TT; t += 256) m = fmaxf(m, s[t]);
    red[tid] = m; __syncthreads();
    for (int st = 128; st > 0; st >>= 1) {
        if (tid < st) red[tid] = fmaxf(red[tid], red[tid + st]);
        __syncthreads();
    }
    m = red[0]; __syncthreads();

    float sum = 0.0f;
    for (int t = tid; t < TT; t += 256) {
        float e = expf(s[t] - m);
        s[t] = e;
        sum += e;
    }
    red[tid] = sum; __syncthreads();
    for (int st = 128; st > 0; st >>= 1) {
        if (tid < st) red[tid] += red[tid + st];
        __syncthreads();
    }
    const float inv = 1.0f / red[0];
    for (int t = tid; t < TT; t += 256) s[t] *= inv;
}

// ---------------------------------------------------------------------------
// Context: partial sums over T-chunks (deterministic, no atomics), then reduce
// ---------------------------------------------------------------------------
__global__ __launch_bounds__(256)
void bahdanau_ctxpart_kernel(const float* __restrict__ values,
                             const float* __restrict__ attn,
                             float* __restrict__ ctxp) {   // [B,8,D]
    const int b = blockIdx.x, tc = blockIdx.y, tid = threadIdx.x;
    __shared__ float wsh[256];
    const float* vb = values + ((size_t)b * TT + tc * 256) * DD;
    const float* ab = attn + (size_t)b * TT + tc * 256;

    wsh[tid] = ab[tid];
    __syncthreads();

    float acc0 = 0.0f, acc1 = 0.0f;
    for (int tt = 0; tt < 256; ++tt) {
        const float w = wsh[tt];
        const float* r = vb + (size_t)tt * DD;
        acc0 += w * r[tid];
        acc1 += w * r[tid + 256];
    }
    float* o = ctxp + ((size_t)b * 8 + tc) * DD;
    o[tid]       = acc0;
    o[tid + 256] = acc1;
}

__global__ void bahdanau_ctxred_kernel(const float* __restrict__ ctxp,
                                       float* __restrict__ ctx) {
    int idx = blockIdx.x * 256 + threadIdx.x;    // B*D threads
    int b = idx / DD, d = idx % DD;
    float acc = 0.0f;
    #pragma unroll
    for (int tc = 0; tc < 8; ++tc)
        acc += ctxp[((size_t)b * 8 + tc) * DD + d];
    ctx[idx] = acc;
}

// ---------------------------------------------------------------------------
extern "C" void kernel_launch(void* const* d_in, const int* in_sizes, int n_in,
                              void* d_out, int out_size, void* d_ws, size_t ws_size,
                              hipStream_t stream) {
    const float* query  = (const float*)d_in[0];   // [B,D]
    const float* values = (const float*)d_in[1];   // [B,T,D]
    const float* W1     = (const float*)d_in[2];   // [D,U]
    const float* b1     = (const float*)d_in[3];   // [U]
    const float* W2     = (const float*)d_in[4];   // [D,U]
    const float* b2     = (const float*)d_in[5];   // [U]
    const float* V      = (const float*)d_in[6];   // [U,1]
    const float* bV     = (const float*)d_in[7];   // [1]

    float* ctx  = (float*)d_out;                   // [B,D]   = 16384 floats
    float* attn = (float*)d_out + BB * DD;         // [B,T]   = 65536 floats

    // Workspace layout: W1t bf16 (512KB) | projq (64KB) | ctx partials (512KB)
    unsigned short* W1t   = (unsigned short*)d_ws;
    float*          projq = (float*)((char*)d_ws + (size_t)UU * DD * sizeof(unsigned short));
    float*          ctxp  = (float*)((char*)projq + (size_t)BB * UU * sizeof(float));

    // 1) Transpose+convert W1 -> bf16 W1t[u][d]
    bahdanau_w1t_kernel<<<(UU * DD) / 256, 256, 0, stream>>>(W1, W1t);
    // 2) projq = query@W2 + b2 + b1 (fp32)
    bahdanau_projq_kernel<<<(BB * UU) / 256, 256, 0, stream>>>(query, W2, b1, b2, projq);
    // 3) Fused WMMA matmul + tanh + dot(V): raw scores into attention region
    bahdanau_score_kernel<<<dim3(TT / 128, BB), 256, 0, stream>>>(
        values, W1t, projq, V, bV, attn);
    // 4) Softmax over T, in place
    bahdanau_softmax_kernel<<<BB, 256, 0, stream>>>(attn);
    // 5) context = sum_t attn * values  (8-way T-split, then serial 8-term reduce)
    bahdanau_ctxpart_kernel<<<dim3(BB, 8), 256, 0, stream>>>(values, attn, ctxp);
    bahdanau_ctxred_kernel<<<(BB * DD) / 256, 256, 0, stream>>>(ctxp, ctx);
}
